// RecencyBiasedAttention_32177894981760
// MI455X (gfx1250) — matrix operand order repaired and checked by reading, not compiled, not run on hardware
//
#include <hip/hip_runtime.h>
#include <hip/hip_bf16.h>
#include <math.h>

// Problem constants (match reference)
#define B_   2
#define S_   4096
#define H_   512
#define NH_  8
#define HD_  64
// log2-domain constants: score*1/sqrt(64)*log2(e), bias (j-i)*ln(0.9)*log2(e)
#define SCALE2 (0.18033688011112042f)   // 0.125 * log2(e)
#define LOGD2  (-0.15200309344504997f)  // ln(0.9) * log2(e)

typedef __attribute__((ext_vector_type(16))) __bf16 v16bf;
typedef __attribute__((ext_vector_type(8)))  float  v8f;

union Frag { v16bf v; unsigned int u[8]; uint4 q[2]; };
union Acc  { v8f   v; float f[8]; };

__device__ __forceinline__ float exp2_hw(float x) {
  return __builtin_amdgcn_exp2f(x);   // v_exp_f32 (hardware is exp2-native)
}
__device__ __forceinline__ unsigned short f32_bf16(float f) {
  union { float f; unsigned int u; } c; c.f = f;
  unsigned int u = c.u;
  return (unsigned short)((u + 0x7FFFu + ((u >> 16) & 1u)) >> 16); // RNE
}
__device__ __forceinline__ unsigned int pk2(float a, float b) {
  return (unsigned int)f32_bf16(a) | ((unsigned int)f32_bf16(b) << 16);
}
__device__ __forceinline__ v8f wmma_bf16(const Frag& a, const Frag& b, v8f c) {
  return __builtin_amdgcn_wmma_f32_16x16x32_bf16(false, a.v, false, b.v,
                                                 (short)0, c, false, false);
}

// ---------------------------------------------------------------------------
// Kernel 0: one-shot f32 -> bf16 conversion (8 elements / thread, vectorized).
// Removes all per-tile re-conversion VALU from the GEMM inner loops.
// ---------------------------------------------------------------------------
__global__ __launch_bounds__(256) void cvt_kernel(
    const float* __restrict__ src, unsigned short* __restrict__ dst)
{
  const size_t i = ((size_t)blockIdx.x * blockDim.x + threadIdx.x) * 8;
  float4 f0 = ((const float4*)(src + i))[0];
  float4 f1 = ((const float4*)(src + i))[1];
  union { uint4 q; unsigned int u[4]; } pk;
  pk.u[0] = pk2(f0.x, f0.y);
  pk.u[1] = pk2(f0.z, f0.w);
  pk.u[2] = pk2(f1.x, f1.y);
  pk.u[3] = pk2(f1.z, f1.w);
  *(uint4*)(dst + i) = pk.q;
}

// ---------------------------------------------------------------------------
// Kernel 1: QKV projection, all-bf16 operands. y[m,o] = sum_h x[m,h]*wq[o,h].
// grid.x = (B*S)/16 token tiles, grid.y = 3 (q/k/v); 8 waves = 8 heads.
// Q,K stored [b][head][s][hd]; V stored transposed [b][head][hd][s].
// Inner loop: 2 global_load_b128 (A) + 8 (B) + 4 wmma per 32-deep K step.
// ---------------------------------------------------------------------------
__global__ __launch_bounds__(256) void qkv_kernel(
    const unsigned short* __restrict__ Xb, const unsigned short* __restrict__ Wq,
    unsigned short* __restrict__ Qb, unsigned short* __restrict__ Kb,
    unsigned short* __restrict__ Vt)
{
  const int m0 = blockIdx.x * 16;
  const int t  = blockIdx.y;                  // 0=Q 1=K 2=V
  const int w  = threadIdx.x >> 5;            // wave id == head
  const int L  = threadIdx.x & 31;
  const int hw = L >> 4, ln = L & 15;
  const int o0 = (t * NH_ + w) * HD_;

  const unsigned short* arow = Xb + (size_t)(m0 + ln) * H_;

  Acc acc[4] = {};
  for (int kk = 0; kk < H_; kk += 32) {
    Frag a;  // A frag: row=ln, lanes<16 K=kk..kk+15, lanes>=16 K=kk+16..31
    {
      const unsigned short* p = arow + kk + (hw ? 16 : 0);
      a.q[0] = *(const uint4*)p;
      a.q[1] = *(const uint4*)(p + 8);
    }
    #pragma unroll
    for (int nt = 0; nt < 4; ++nt) {
      const unsigned short* wr = Wq + (size_t)(o0 + nt * 16 + ln) * H_ + kk + (hw ? 16 : 0);
      __builtin_prefetch(wr + 32, 0, 1);
      Frag bf;  // B frag: col=ln, lanes<16 K=kk..kk+15, lanes>=16 K=kk+16..31
      bf.q[0] = ((const uint4*)wr)[0];
      bf.q[1] = ((const uint4*)wr)[1];
      acc[nt].v = wmma_bf16(a, bf, acc[nt].v);
    }
  }

  const int bb = m0 >> 12;                    // batch (S_ = 4096)
  const int s0 = m0 & (S_ - 1);
  if (t < 2) {
    unsigned short* dst = (t == 0 ? Qb : Kb) + ((size_t)bb * NH_ + w) * S_ * HD_;
    #pragma unroll
    for (int nt = 0; nt < 4; ++nt) {
      const int d = nt * 16 + ln;
      #pragma unroll
      for (int r = 0; r < 8; ++r)
        dst[(size_t)(s0 + r + 8 * hw) * HD_ + d] = f32_bf16(acc[nt].f[r]);
    }
  } else {
    // V transposed: per lane 8 consecutive tokens at fixed hd -> one 16B store
    unsigned short* dst = Vt + ((size_t)bb * NH_ + w) * HD_ * S_;
    #pragma unroll
    for (int nt = 0; nt < 4; ++nt) {
      const int d = nt * 16 + ln;
      union { uint4 q; unsigned int u[4]; } pk;
      #pragma unroll
      for (int i = 0; i < 4; ++i)
        pk.u[i] = pk2(acc[nt].f[2 * i], acc[nt].f[2 * i + 1]);
      *(uint4*)&dst[(size_t)d * S_ + s0 + 8 * hw] = pk.q;
    }
  }
}

// ---------------------------------------------------------------------------
// Kernel 2: flash-style attention with recency bias (online softmax, log2
// domain so v_exp_f32 is used directly). grid.x = B*NH, grid.y = S/128;
// each wave owns a 16-query tile, keys consumed 32 at a time (8 wmma/chunk).
// ---------------------------------------------------------------------------
__global__ __launch_bounds__(256) void attn_kernel(
    const unsigned short* __restrict__ Qb, const unsigned short* __restrict__ Kb,
    const unsigned short* __restrict__ Vt, unsigned short* __restrict__ Ob)
{
  __shared__ __align__(16) unsigned short ps[8][16][40];  // per-wave P tile
  const int bh = blockIdx.x;
  const int bb = bh >> 3, h = bh & 7;
  const int w  = threadIdx.x >> 5;
  const int L  = threadIdx.x & 31;
  const int hw = L >> 4, ln = L & 15;
  const int q0 = blockIdx.y * 128 + w * 16;

  const unsigned short* Q = Qb + (size_t)bh * S_ * HD_;
  const unsigned short* K = Kb + (size_t)bh * S_ * HD_;
  const unsigned short* V = Vt + (size_t)bh * HD_ * S_;

  // Q A-frags (hd 0..31 / 32..63) pinned in registers for the whole key loop
  Frag qa0, qa1;
  {
    const unsigned short* p = Q + (size_t)(q0 + ln) * HD_ + (hw ? 16 : 0);
    qa0.q[0] = *(const uint4*)p;          qa0.q[1] = *(const uint4*)(p + 8);
    qa1.q[0] = *(const uint4*)(p + 32);   qa1.q[1] = *(const uint4*)(p + 40);
  }

  float mrow[8], lrow[8];
  Acc o[4] = {};
  #pragma unroll
  for (int r = 0; r < 8; ++r) { mrow[r] = -1e30f; lrow[r] = 0.0f; }

  for (int kc = 0; kc < S_; kc += 32) {
    Frag kb0l, kb0h, kb1l, kb1h;
    {
      const unsigned short* p0 = K + (size_t)(kc + ln) * HD_ + (hw ? 16 : 0);
      kb0l.q[0] = ((const uint4*)p0)[0];        kb0l.q[1] = ((const uint4*)p0)[1];
      kb0h.q[0] = ((const uint4*)(p0 + 32))[0]; kb0h.q[1] = ((const uint4*)(p0 + 32))[1];
      const unsigned short* p1 = p0 + 16 * HD_;
      kb1l.q[0] = ((const uint4*)p1)[0];        kb1l.q[1] = ((const uint4*)p1)[1];
      kb1h.q[0] = ((const uint4*)(p1 + 32))[0]; kb1h.q[1] = ((const uint4*)(p1 + 32))[1];
    }
    v8f z = {};
    Acc s0c, s1c;
    s0c.v = wmma_bf16(qa0, kb0l, z);
    s0c.v = wmma_bf16(qa1, kb0h, s0c.v);
    s1c.v = wmma_bf16(qa0, kb1l, z);
    s1c.v = wmma_bf16(qa1, kb1h, s1c.v);

    // Online softmax on this 16x32 chunk (log2 domain)
    #pragma unroll
    for (int r = 0; r < 8; ++r) {
      const int i  = q0 + r + 8 * hw;
      const int j0 = kc + ln;
      const int j1 = j0 + 16;
      float sc0 = s0c.f[r] * SCALE2 + (j0 <= i ? (float)(j0 - i) * LOGD2 : 0.0f);
      float sc1 = s1c.f[r] * SCALE2 + (j1 <= i ? (float)(j1 - i) * LOGD2 : 0.0f);
      float vmax = fmaxf(sc0, sc1);
      #pragma unroll
      for (int d = 1; d < 16; d <<= 1) vmax = fmaxf(vmax, __shfl_xor(vmax, d, 32));
      const float mn    = fmaxf(mrow[r], vmax);
      const float alpha = exp2_hw(mrow[r] - mn);
      mrow[r] = mn;
      const float p0 = exp2_hw(sc0 - mn);
      const float p1 = exp2_hw(sc1 - mn);
      float rs = p0 + p1;
      #pragma unroll
      for (int d = 1; d < 16; d <<= 1) rs += __shfl_xor(rs, d, 32);
      lrow[r] = lrow[r] * alpha + rs;
      #pragma unroll
      for (int nt = 0; nt < 4; ++nt) o[nt].f[r] *= alpha;
      ps[w][r + 8 * hw][ln]      = f32_bf16(p0);   // D-layout -> LDS
      ps[w][r + 8 * hw][16 + ln] = f32_bf16(p1);
    }

    // P back as A-frag (16x32) via per-wave LDS transpose
    Frag pa;
    {
      const unsigned short* p = &ps[w][ln][hw ? 16 : 0];
      pa.q[0] = *(const uint4*)p;
      pa.q[1] = *(const uint4*)(p + 8);
    }
    // V B-frags from transposed V (contiguous along keys)
    #pragma unroll
    for (int nt = 0; nt < 4; ++nt) {
      const unsigned short* vp = V + (size_t)(nt * 16 + ln) * S_ + kc + (hw ? 16 : 0);
      Frag vb;
      vb.q[0] = ((const uint4*)vp)[0];
      vb.q[1] = ((const uint4*)vp)[1];
      o[nt].v = wmma_bf16(pa, vb, o[nt].v);
    }
  }

  // Normalize and emit O in [b][s][head*64+d] layout (bf16)
  #pragma unroll
  for (int nt = 0; nt < 4; ++nt)
    #pragma unroll
    for (int r = 0; r < 8; ++r) {
      const int tok = q0 + r + 8 * hw;
      Ob[((size_t)bb * S_ + tok) * H_ + h * HD_ + nt * 16 + ln] =
          f32_bf16(o[nt].f[r] / lrow[r]);
    }
}

// ---------------------------------------------------------------------------
// Kernel 3: output projection y = O @ w_out^T + b_out (f32 out), all-bf16
// operands in the inner loop. grid.x = (B*S)/16; 8 waves cover 512 columns.
// ---------------------------------------------------------------------------
__global__ __launch_bounds__(256) void proj_kernel(
    const unsigned short* __restrict__ Ob, const unsigned short* __restrict__ Wo,
    const float* __restrict__ b_out, float* __restrict__ out)
{
  const int m0 = blockIdx.x * 16;
  const int w  = threadIdx.x >> 5;
  const int L  = threadIdx.x & 31;
  const int hw = L >> 4, ln = L & 15;
  const int n0 = w * 64;

  const unsigned short* arow = Ob + (size_t)(m0 + ln) * H_;

  Acc acc[4] = {};
  for (int kk = 0; kk < H_; kk += 32) {
    Frag a;
    const unsigned short* p = arow + kk + (hw ? 16 : 0);
    a.q[0] = *(const uint4*)p;
    a.q[1] = *(const uint4*)(p + 8);
    #pragma unroll
    for (int nt = 0; nt < 4; ++nt) {
      const unsigned short* wr = Wo + (size_t)(n0 + nt * 16 + ln) * H_ + kk + (hw ? 16 : 0);
      __builtin_prefetch(wr + 32, 0, 1);
      Frag bf;
      bf.q[0] = ((const uint4*)wr)[0];
      bf.q[1] = ((const uint4*)wr)[1];
      acc[nt].v = wmma_bf16(a, bf, acc[nt].v);
    }
  }
  #pragma unroll
  for (int nt = 0; nt < 4; ++nt) {
    const int oc = n0 + nt * 16 + ln;
    const float bias = b_out[oc];
    #pragma unroll
    for (int r = 0; r < 8; ++r)
      out[(size_t)(m0 + r + 8 * hw) * H_ + oc] = acc[nt].f[r] + bias;
  }
}

// ---------------------------------------------------------------------------
extern "C" void kernel_launch(void* const* d_in, const int* in_sizes, int n_in,
                              void* d_out, int out_size, void* d_ws, size_t ws_size,
                              hipStream_t stream) {
  (void)in_sizes; (void)n_in; (void)out_size; (void)ws_size;
  const float* x     = (const float*)d_in[0];
  const float* w_qkv = (const float*)d_in[1];
  const float* w_out = (const float*)d_in[2];
  const float* b_out = (const float*)d_in[3];
  float* out = (float*)d_out;

  // Workspace layout (bf16 buffers, elements):
  //   Xb 4Mi | Wq 768Ki | Wo 256Ki | Qb 4Mi | Kb 4Mi | Vt 4Mi | Ob 4Mi  (~42MB)
  const size_t nX = (size_t)B_ * S_ * H_;         // 4Mi
  const size_t nWq = (size_t)3 * H_ * H_;         // 768Ki
  const size_t nWo = (size_t)H_ * H_;             // 256Ki
  const size_t nE  = (size_t)B_ * NH_ * S_ * HD_; // 4Mi
  unsigned short* Xb = (unsigned short*)d_ws;
  unsigned short* Wq = Xb + nX;
  unsigned short* Wo = Wq + nWq;
  unsigned short* Qb = Wo + nWo;
  unsigned short* Kb = Qb + nE;
  unsigned short* Vt = Kb + nE;
  unsigned short* Ob = Vt + nE;

  dim3 blk(256);
  cvt_kernel<<<dim3((unsigned)(nX  / 2048)), blk, 0, stream>>>(x, Xb);
  cvt_kernel<<<dim3((unsigned)(nWq / 2048)), blk, 0, stream>>>(w_qkv, Wq);
  cvt_kernel<<<dim3((unsigned)(nWo / 2048)), blk, 0, stream>>>(w_out, Wo);
  qkv_kernel<<<dim3((B_ * S_) / 16, 3), blk, 0, stream>>>(Xb, Wq, Qb, Kb, Vt);
  attn_kernel<<<dim3(B_ * NH_, S_ / 128), blk, 0, stream>>>(Qb, Kb, Vt, Ob);
  proj_kernel<<<dim3((B_ * S_) / 16), blk, 0, stream>>>(Ob, Wo, b_out, out);
}
